// NodeEdgeEarlyInteraction_72533407694905
// MI455X (gfx1250) — matrix operand order repaired and checked by reading, not compiled, not run on hardware
//
#include <hip/hip_runtime.h>
#include <hip/hip_bf16.h>

// Problem constants (match reference)
#define BB 256
#define MAXSET 64
#define EE 512
#define DD 128
#define NPROP 5
#define NTIME 3
#define NGg (2*BB)          // 512 graphs
#define NNn (NGg*MAXSET)    // 32768 nodes
#define TOTE (NGg*EE)       // 262144 edges

typedef unsigned short bf16_t;
typedef __attribute__((ext_vector_type(16))) __bf16 v16bf;
typedef __attribute__((ext_vector_type(8)))  float  v8f;

union FragU { uint4 u[2]; v16bf v; };

__device__ inline bf16_t f2bf(float f){
  unsigned int u = __float_as_uint(f);
  unsigned int r = (u + 0x7FFFu + ((u >> 16) & 1u)) >> 16;  // RNE
  return (bf16_t)r;
}
__device__ inline float bf2f(bf16_t h){
  return __uint_as_float(((unsigned int)h) << 16);
}

__device__ inline v8f vzero8(){
  v8f z;
#pragma unroll
  for (int i = 0; i < 8; ++i) z[i] = 0.0f;
  return z;
}

// A-fragment (16x32 bf16) from an LDS row-major tile: 2x ds_read_b128.
__device__ inline v16bf frag_a_lds(const bf16_t* base, int stride, int row, int kbase){
  int hi = (threadIdx.x >> 4) & 1;
  const bf16_t* p = base + row * stride + kbase + hi * 8;
  FragU f;
  f.u[0] = *(const uint4*)(p);
  f.u[1] = *(const uint4*)(p + 16);
  return f.v;
}

// B-fragment (32x16 bf16) from pre-packed weights: 32B contiguous per lane.
__device__ inline v16bf frag_b_packed(const bf16_t* wp, int KC, int ntile, int kc){
  int lane = threadIdx.x & 31;
  const bf16_t* p = wp + ((size_t)(((ntile * KC) + kc) * 32 + lane) << 4);
  FragU f;
  f.u[0] = *(const uint4*)(p);
  f.u[1] = *(const uint4*)(p + 8);
  return f.v;
}

// ---------------------------------------------------------------------------
// Tensor Data Mover: async 2D bf16 tile load (global -> LDS).
// D# group0: count=1 | lds_addr | global_addr | type=2.
// D# group1: data_size=1(2B) | tensor_dim0/1 | tile_dim0/1 | dim0 stride.
// ---------------------------------------------------------------------------
#if __has_builtin(__builtin_amdgcn_tensor_load_to_lds)
#define USE_TDM 1
#else
#define USE_TDM 0
#endif

#if USE_TDM
typedef __attribute__((ext_vector_type(4))) unsigned int v4u_t;
typedef __attribute__((ext_vector_type(4))) int v4i_t;
typedef __attribute__((ext_vector_type(8))) int v8i_t;

__device__ inline void tdm_load_tile_bf16(unsigned lds_byte_off, const bf16_t* gsrc,
                                          unsigned tile_w, unsigned tile_h,
                                          unsigned row_stride_elems){
  unsigned long long ga = (unsigned long long)(size_t)gsrc;
  v4u_t g0;
  g0[0] = 1u;                                                  // count=1 (user D#)
  g0[1] = lds_byte_off;                                        // lds_addr
  g0[2] = (unsigned)ga;                                        // global_addr[31:0]
  g0[3] = (unsigned)((ga >> 32) & 0x01FFFFFFu) | 0x80000000u;  // ga[56:32] | type=2
  union { unsigned long long q[4]; v8i_t v; } g1;
  unsigned long long td0 = row_stride_elems;                   // tensor_dim0
  unsigned long long td1 = 1ull << 20;                         // tensor_dim1 (ample)
  g1.q[0] = (1ull << 16) | (td0 << 48);                        // data_size=2B; dim0 lo16
  g1.q[1] = (td0 >> 16) | (td1 << 16) | ((unsigned long long)tile_w << 48);
  g1.q[2] = (unsigned long long)tile_h | ((unsigned long long)row_stride_elems << 32);
  g1.q[3] = 0ull;                                              // dim1 stride unused (2D)
  v4i_t z4; z4[0] = 0; z4[1] = 0; z4[2] = 0; z4[3] = 0;
#if __clang_major__ >= 23
  v8i_t z8;
#pragma unroll
  for (int i = 0; i < 8; ++i) z8[i] = 0;
  __builtin_amdgcn_tensor_load_to_lds(g0, g1.v, z4, z4, z8, 0);
#else
  __builtin_amdgcn_tensor_load_to_lds(g0, g1.v, z4, z4, 0);
#endif
}

__device__ inline void tdm_wait(){
#if __has_builtin(__builtin_amdgcn_s_wait_tensorcnt)
  __builtin_amdgcn_s_wait_tensorcnt(0);
#else
  asm volatile("s_wait_tensorcnt 0x0" ::: "memory");
#endif
}
#endif  // USE_TDM

// Manual fallback: contiguous-row bf16 tile copy (uint4 granules).
__device__ inline void copy_tile_bf16(bf16_t* dst, const bf16_t* src, int row0,
                                      int stride_elems, int w_u4 /* tile_w/8 */){
  for (int i = threadIdx.x; i < 64 * w_u4; i += 256){
    int r = i / w_u4, q = i - r * w_u4;
    ((uint4*)dst)[i] = ((const uint4*)(src + (size_t)(row0 + r) * stride_elems))[q];
  }
}

// ---------------------------------------------------------------------------
// Weight repack: fp32 W[K x N] -> bf16 WMMA B-fragments, zero-padded in K.
// ---------------------------------------------------------------------------
__global__ void k_repack(const float* __restrict__ W, bf16_t* __restrict__ out,
                         int Kreal, int Ncols, int KC){
  int t = blockIdx.x * blockDim.x + threadIdx.x;
  int total = (Ncols / 16) * KC * 32;
  if (t >= total) return;
  int lane  = t & 31;
  int kc    = (t >> 5) % KC;
  int ntile = (t >> 5) / KC;
  int n   = ntile * 16 + (lane & 15);
  int khi = (lane >> 4) * 8;
  bf16_t* o = out + (size_t)t * 16;
#pragma unroll
  for (int j = 0; j < 16; ++j){
    int k = kc * 32 + khi + j + ((j >> 3) << 3);
    float v = (k < Kreal) ? W[(size_t)k * Ncols + n] : 0.0f;
    o[j] = f2bf(v);
  }
}

__global__ void k_zero(float* __restrict__ p, long long n){
  long long i = (long long)blockIdx.x * blockDim.x + threadIdx.x;
  if (i < n) p[i] = 0.0f;
}

// ---------------------------------------------------------------------------
// Single-layer WMMA linear (fp32 in, bf16 out): encoders.
// ---------------------------------------------------------------------------
template<int K1, int NOUT>
__global__ __launch_bounds__(256) void k_linear(const float* __restrict__ X, int xstride, int kreal,
                                                const bf16_t* __restrict__ Wp,
                                                const float* __restrict__ bias,
                                                bf16_t* __restrict__ Y, int ystride){
  constexpr int KC = K1 / 32;
  __shared__ __attribute__((aligned(16))) bf16_t sX[64 * K1];
  int row0 = blockIdx.x * 64;
  for (int i = threadIdx.x; i < 64 * K1; i += 256){
    int r = i / K1, c = i - r * K1;
    float v = (c < kreal) ? X[(size_t)(row0 + r) * xstride + c] : 0.0f;
    sX[i] = f2bf(v);
  }
  __syncthreads();
  int w = threadIdx.x >> 5, lane = threadIdx.x & 31;
  int mtile = w & 3;
  constexpr int NTW = NOUT / 32;
  int nbase = (w >> 2) * NTW;
  v8f acc[NTW];
#pragma unroll
  for (int nt = 0; nt < NTW; ++nt) acc[nt] = vzero8();
  int arow = mtile * 16 + (lane & 15);
#pragma unroll
  for (int kc = 0; kc < KC; ++kc){
    v16bf a = frag_a_lds(sX, K1, arow, kc * 32);
#pragma unroll
    for (int nt = 0; nt < NTW; ++nt){
      v16bf b = frag_b_packed(Wp, KC, nbase + nt, kc);
      acc[nt] = __builtin_amdgcn_wmma_f32_16x16x32_bf16(false, a, false, b,
                                                        (short)0, acc[nt], false, false);
    }
  }
  int rlo = (lane >> 4) * 8, c0 = lane & 15;
#pragma unroll
  for (int nt = 0; nt < NTW; ++nt){
    int col = (nbase + nt) * 16 + c0;
    float bv = bias[col];
    int off0 = (row0 + mtile * 16 + rlo) * ystride + col;
#pragma unroll
    for (int r = 0; r < 8; ++r)
      Y[off0 + r * ystride] = f2bf(acc[nt][r] + bv);
  }
}

// ---------------------------------------------------------------------------
// Fused 2-layer MLP:  Y = relu(X @ W1 + b1) @ W2 + b2
// All activations bf16; WMMA accumulates f32; hidden tile bf16 in LDS.
// GATHER: 0 = one bf16 src (TDM)            K1=128
//         1 = two bf16 srcs (2x TDM)        K1=256
//         2 = msg: comb[from]|comb[to]|e0   K1=320 (manual row gather)
//         3 = bf16 src (TDM) | f32 src      K1=256 (node MLP: comb|agg)
// SCATTER: atomicAdd f32 into Yf[to_idx[row]*ystride + col]
// LDS X layout: seg0[64x128] | seg1[64x128] | seg2[64x64]; chunks never cross.
// ---------------------------------------------------------------------------
template<int K1, int HID, int NOUT, int GATHER, bool SCATTER>
__global__ __launch_bounds__(256) void k_mlp2(
    const bf16_t* __restrict__ s0, int s0stride,
    const bf16_t* __restrict__ s1bf, int s1stride,
    const float* __restrict__ s1f32,
    const bf16_t* __restrict__ e0,
    const int* __restrict__ from_idx, const int* __restrict__ to_idx,
    const bf16_t* __restrict__ W1p, const float* __restrict__ b1,
    const bf16_t* __restrict__ W2p, const float* __restrict__ b2,
    float* __restrict__ Yf, bf16_t* __restrict__ Ybf, int ystride){
  constexpr int KC1 = K1 / 32;
  constexpr int KC2 = HID / 32;
  __shared__ __attribute__((aligned(16))) bf16_t sX[64 * K1];
  __shared__ __attribute__((aligned(16))) bf16_t sH[64 * HID];
  int row0 = blockIdx.x * 64;

  // ---- Stage 0: gather input tile into LDS (bf16) ----
  if (GATHER == 0 || GATHER == 1 || GATHER == 3){
    int seg0w = (K1 == 128) ? 128 : 128;   // seg0 is always 128 wide here
#if USE_TDM
    if (threadIdx.x == 0){
      unsigned lds0 = (unsigned)(size_t)(void*)sX;
      tdm_load_tile_bf16(lds0, s0 + (size_t)row0 * s0stride, seg0w, 64, s0stride);
      if (GATHER == 1)
        tdm_load_tile_bf16(lds0 + 64u * 128u * 2u, s1bf + (size_t)row0 * s1stride,
                           128, 64, s1stride);
    }
#else
    copy_tile_bf16(sX, s0, row0, s0stride, seg0w / 8);
    if (GATHER == 1)
      copy_tile_bf16(sX + 64 * 128, s1bf, row0, s1stride, 16);
#endif
    if (GATHER == 3){   // second source: f32 agg -> bf16 (manual)
      for (int i = threadIdx.x; i < 64 * 128; i += 256){
        int r = i >> 7, c = i & 127;
        sX[64 * 128 + i] = f2bf(s1f32[(size_t)(row0 + r) * 128 + c]);
      }
    }
#if USE_TDM
    tdm_wait();
#endif
  } else {  // GATHER == 2: row-gathered message input
    for (int i = threadIdx.x; i < 64 * 16; i += 256){   // seg0/seg1: 16 uint4 per row
      int r = i >> 4, q = i & 15;
      int er = row0 + r;
      ((uint4*)sX)[i]            = ((const uint4*)(s0 + (size_t)from_idx[er] * s0stride))[q];
      ((uint4*)(sX + 64 * 128))[i] = ((const uint4*)(s0 + (size_t)to_idx[er] * s0stride))[q];
    }
    for (int i = threadIdx.x; i < 64 * 8; i += 256){    // seg2: e0, 8 uint4 per row
      int r = i >> 3, q = i & 7;
      ((uint4*)(sX + 2 * 64 * 128))[i] = ((const uint4*)(e0 + (size_t)(row0 + r) * 64))[q];
    }
  }
  __syncthreads();

  int w = threadIdx.x >> 5, lane = threadIdx.x & 31;
  int mtile = w & 3;
  int rlo = (lane >> 4) * 8, c0 = lane & 15;
  int arow = mtile * 16 + (lane & 15);

  // ---- Stage 1: hidden = relu(X @ W1 + b1) -> LDS bf16 ----
  {
    constexpr int NTW = HID / 32;
    int nbase = (w >> 2) * NTW;
    v8f acc[NTW];
#pragma unroll
    for (int nt = 0; nt < NTW; ++nt) acc[nt] = vzero8();
#pragma unroll
    for (int kc = 0; kc < KC1; ++kc){
      const bf16_t* base; int stride, kb;
      if (K1 == 128){ base = sX; stride = 128; kb = kc * 32; }
      else if (kc < 4){ base = sX; stride = 128; kb = kc * 32; }
      else if (kc < 8){ base = sX + 64 * 128; stride = 128; kb = (kc - 4) * 32; }
      else            { base = sX + 2 * 64 * 128; stride = 64; kb = (kc - 8) * 32; }
      v16bf a = frag_a_lds(base, stride, arow, kb);
#pragma unroll
      for (int nt = 0; nt < NTW; ++nt){
        v16bf b = frag_b_packed(W1p, KC1, nbase + nt, kc);
        acc[nt] = __builtin_amdgcn_wmma_f32_16x16x32_bf16(false, a, false, b,
                                                          (short)0, acc[nt], false, false);
      }
    }
#pragma unroll
    for (int nt = 0; nt < NTW; ++nt){
      int col = (nbase + nt) * 16 + c0;
      float bv = b1[col];
#pragma unroll
      for (int r = 0; r < 8; ++r){
        float v = acc[nt][r] + bv;
        v = v > 0.0f ? v : 0.0f;
        sH[(mtile * 16 + rlo + r) * HID + col] = f2bf(v);
      }
    }
  }
  __syncthreads();

  // ---- Stage 2: out = hidden @ W2 + b2 ----
  {
    constexpr int NTW = NOUT / 32;
    int nbase = (w >> 2) * NTW;
    v8f acc[NTW];
#pragma unroll
    for (int nt = 0; nt < NTW; ++nt) acc[nt] = vzero8();
#pragma unroll
    for (int kc = 0; kc < KC2; ++kc){
      v16bf a = frag_a_lds(sH, HID, arow, kc * 32);
#pragma unroll
      for (int nt = 0; nt < NTW; ++nt){
        v16bf b = frag_b_packed(W2p, KC2, nbase + nt, kc);
        acc[nt] = __builtin_amdgcn_wmma_f32_16x16x32_bf16(false, a, false, b,
                                                          (short)0, acc[nt], false, false);
      }
    }
#pragma unroll
    for (int nt = 0; nt < NTW; ++nt){
      int col = (nbase + nt) * 16 + c0;
      float bv = b2[col];
      int rowb = row0 + mtile * 16 + rlo;
#pragma unroll
      for (int r = 0; r < 8; ++r){
        float v = acc[nt][r] + bv;
        if (SCATTER){
          atomicAdd(Yf + (size_t)to_idx[rowb + r] * ystride + col, v);
        } else {
          Ybf[(rowb + r) * ystride + col] = f2bf(v);
        }
      }
    }
  }
}

// ---------------------------------------------------------------------------
// Per-pair similarity + Sinkhorn (bf16 t-features in, f32 plan out).
// ---------------------------------------------------------------------------
__global__ __launch_bounds__(256) void k_simsink(const bf16_t* __restrict__ tAll,
                                                 float* __restrict__ plan){
  __shared__ float la[64 * 64];
  int b = blockIdx.x;
  const bf16_t* tq = tAll + (size_t)(2 * b)     * 64 * 64;
  const bf16_t* tc = tAll + (size_t)(2 * b + 1) * 64 * 64;
  const float invT = 10.0f;  // 1/TEMP
  for (int idx = threadIdx.x; idx < 4096; idx += 256){
    int i = idx >> 6, j = idx & 63;
    float d = 0.0f;
    for (int k = 0; k < 64; ++k) d += bf2f(tq[i * 64 + k]) * bf2f(tc[j * 64 + k]);
    la[idx] = d * invT;
  }
  for (int it = 0; it < 20; ++it){
    __syncthreads();
    if (threadIdx.x < 64){            // normalize over columns (axis=2)
      int i = threadIdx.x;
      float m = -3.0e38f;
      for (int j = 0; j < 64; ++j) m = fmaxf(m, la[i * 64 + j]);
      float s = 0.0f;
      for (int j = 0; j < 64; ++j) s += __expf(la[i * 64 + j] - m);
      float l = m + __logf(s);
      for (int j = 0; j < 64; ++j) la[i * 64 + j] -= l;
    }
    __syncthreads();
    if (threadIdx.x < 64){            // normalize over rows (axis=1)
      int j = threadIdx.x;
      float m = -3.0e38f;
      for (int i = 0; i < 64; ++i) m = fmaxf(m, la[i * 64 + j]);
      float s = 0.0f;
      for (int i = 0; i < 64; ++i) s += __expf(la[i * 64 + j] - m);
      float l = m + __logf(s);
      for (int i = 0; i < 64; ++i) la[i * 64 + j] -= l;
    }
  }
  __syncthreads();
  for (int idx = threadIdx.x; idx < 4096; idx += 256)
    plan[(size_t)b * 4096 + idx] = __expf(la[idx]);
}

// ---------------------------------------------------------------------------
// Cross-graph mixing: qfc = plan @ c, cfq = plan^T @ q (cols D:768),
// SB (bf16 parts) -> SA (bf16 store; cols 0:D stay zero).
// ---------------------------------------------------------------------------
__global__ __launch_bounds__(256) void k_mix(const float* __restrict__ plan,
                                             const bf16_t* __restrict__ SB,
                                             bf16_t* __restrict__ SA){
  __shared__ float P[64 * 64];
  int b = blockIdx.x;
  for (int i = threadIdx.x; i < 4096; i += 256) P[i] = plan[(size_t)b * 4096 + i];
  __syncthreads();
  const bf16_t* qrow = SB + (size_t)(2 * b)     * 64 * 768;
  const bf16_t* crow = SB + (size_t)(2 * b + 1) * 64 * 768;
  bf16_t* oq = SA + (size_t)(2 * b)     * 64 * 768;
  bf16_t* oc = SA + (size_t)(2 * b + 1) * 64 * 768;
  for (int idx = threadIdx.x; idx < 64 * 640; idx += 256){
    int i = idx / 640, j = idx - i * 640;
    float a0 = 0.0f, a1 = 0.0f;
    for (int k = 0; k < 64; ++k){
      a0 += P[i * 64 + k] * bf2f(crow[k * 768 + 128 + j]);   // qfc
      a1 += P[k * 64 + i] * bf2f(qrow[k * 768 + 128 + j]);   // cfq
    }
    oq[i * 768 + 128 + j] = f2bf(a0);
    oc[i * 768 + 128 + j] = f2bf(a1);
  }
}

// ---------------------------------------------------------------------------
extern "C" void kernel_launch(void* const* d_in, const int* in_sizes, int n_in,
                              void* d_out, int out_size, void* d_ws, size_t ws_size,
                              hipStream_t stream){
  const float* node_features = (const float*)d_in[0];
  const float* edge_features = (const float*)d_in[1];
  const int*   from_idx      = (const int*)d_in[2];
  const int*   to_idx        = (const int*)d_in[3];
  const float* node_enc_w = (const float*)d_in[4];
  const float* node_enc_b = (const float*)d_in[5];
  const float* edge_enc_w = (const float*)d_in[6];
  const float* edge_enc_b = (const float*)d_in[7];
  const float* mw1 = (const float*)d_in[8];
  const float* mb1 = (const float*)d_in[9];
  const float* mw2 = (const float*)d_in[10];
  const float* mb2 = (const float*)d_in[11];
  const float* nw1 = (const float*)d_in[12];
  const float* nb1 = (const float*)d_in[13];
  const float* nw2 = (const float*)d_in[14];
  const float* nb2 = (const float*)d_in[15];
  const float* cw1 = (const float*)d_in[16];
  const float* cb1 = (const float*)d_in[17];
  const float* cw2 = (const float*)d_in[18];
  const float* cb2 = (const float*)d_in[19];
  const float* t1w = (const float*)d_in[20];
  const float* t1b = (const float*)d_in[21];
  const float* t2w = (const float*)d_in[22];
  const float* t2b = (const float*)d_in[23];

  char* base = (char*)d_ws;
  size_t off = 0;
  auto take = [&](size_t bytes) -> char* {
    char* p = base + off;
    off = (off + bytes + 255) & ~(size_t)255;
    return p;
  };
  bf16_t* h0   = (bf16_t*)take((size_t)NNn  * 128 * 2);
  bf16_t* e0   = (bf16_t*)take((size_t)TOTE * 64  * 2);
  bf16_t* SA   = (bf16_t*)take((size_t)NNn  * 768 * 2);  // store: zeros + mixed
  bf16_t* SB   = (bf16_t*)take((size_t)NNn  * 768 * 2);  // fresh parts
  bf16_t* comb = (bf16_t*)take((size_t)NNn  * 128 * 2);
  float*  agg  = (float*) take((size_t)NNn  * 128 * 4);  // f32 atomic accumulator
  bf16_t* tAll = (bf16_t*)take((size_t)NNn  * 64  * 2);
  auto takew = [&](int Kreal, int Ncols) -> bf16_t* {
    int KC = (Kreal + 31) / 32;
    return (bf16_t*)take((size_t)(Ncols / 16) * KC * 32 * 16 * 2);
  };
  bf16_t* pw_node = takew(32, 128);
  bf16_t* pw_edge = takew(16, 64);
  bf16_t* pw_cw1  = takew(256, 256);
  bf16_t* pw_cw2  = takew(256, 128);
  bf16_t* pw_mw1  = takew(320, 256);
  bf16_t* pw_mw2  = takew(256, 128);
  bf16_t* pw_nw1  = takew(256, 256);
  bf16_t* pw_nw2  = takew(256, 128);
  bf16_t* pw_t1   = takew(128, 64);
  bf16_t* pw_t2   = takew(64, 64);

  auto repack = [&](const float* W, bf16_t* outp, int Kreal, int Ncols){
    int KC = (Kreal + 31) / 32;
    int total = (Ncols / 16) * KC * 32;
    k_repack<<<dim3((total + 255) / 256), dim3(256), 0, stream>>>(W, outp, Kreal, Ncols, KC);
  };
  repack(node_enc_w, pw_node, 32, 128);
  repack(edge_enc_w, pw_edge, 16, 64);
  repack(cw1, pw_cw1, 256, 256);
  repack(cw2, pw_cw2, 256, 128);
  repack(mw1, pw_mw1, 320, 256);
  repack(mw2, pw_mw2, 256, 128);
  repack(nw1, pw_nw1, 256, 256);
  repack(nw2, pw_nw2, 256, 128);
  repack(t1w, pw_t1, 128, 64);
  repack(t2w, pw_t2, 64, 64);

  // store := zeros every call (graph-replay determinism); bf16 zero == 0x0000
  {
    long long n = (long long)NNn * 768 / 2;   // bf16 elements as f32 words
    k_zero<<<dim3((unsigned)((n + 255) / 256)), dim3(256), 0, stream>>>((float*)SA, n);
  }

  // Encoders (fp32 features -> bf16 activations)
  k_linear<32, 128><<<dim3(NNn / 64), dim3(256), 0, stream>>>(
      node_features, 32, 32, pw_node, node_enc_b, h0, 128);
  k_linear<32, 64><<<dim3(TOTE / 64), dim3(256), 0, stream>>>(
      edge_features, 16, 16, pw_edge, edge_enc_b, e0, 64);

  float* plan = (float*)d_out;

  for (int t = 0; t < NTIME; ++t){
    for (int p = 1; p <= NPROP; ++p){
      const bf16_t* hsrc = (p == 1) ? h0 : (SB + (size_t)(p - 1) * 128);
      int hstride = (p == 1) ? 128 : 768;
      // comb = MLP_c([h, inter])   (two bf16 tiles via TDM)
      k_mlp2<256, 256, 128, 1, false><<<dim3(NNn / 64), dim3(256), 0, stream>>>(
          hsrc, hstride, SA + (size_t)(p - 1) * 128, 768, nullptr,
          nullptr, nullptr, nullptr,
          pw_cw1, cb1, pw_cw2, cb2, nullptr, comb, 128);
      // agg = 0
      {
        long long n = (long long)NNn * 128;
        k_zero<<<dim3((unsigned)((n + 255) / 256)), dim3(256), 0, stream>>>(agg, n);
      }
      // messages = MLP_m([comb[from], comb[to], e0]); agg += scatter(to_idx)
      k_mlp2<320, 256, 128, 2, true><<<dim3(TOTE / 64), dim3(256), 0, stream>>>(
          comb, 128, nullptr, 0, nullptr,
          e0, from_idx, to_idx,
          pw_mw1, mb1, pw_mw2, mb2, agg, nullptr, 128);
      // h = MLP_n([comb, agg]) -> SB part p  (comb via TDM, agg f32 converted)
      k_mlp2<256, 256, 128, 3, false><<<dim3(NNn / 64), dim3(256), 0, stream>>>(
          comb, 128, nullptr, 0, agg,
          nullptr, nullptr, nullptr,
          pw_nw1, nb1, pw_nw2, nb2, nullptr, SB + (size_t)p * 128, 768);
    }
    // tAll = MLP_t(last part)  (single bf16 tile via TDM)
    k_mlp2<128, 64, 64, 0, false><<<dim3(NNn / 64), dim3(256), 0, stream>>>(
        SB + 640, 768, nullptr, 0, nullptr,
        nullptr, nullptr, nullptr,
        pw_t1, t1b, pw_t2, t2b, nullptr, tAll, 64);
    // sim + sinkhorn -> plan (d_out); mix parts back into SA cols D:768
    k_simsink<<<dim3(BB), dim3(256), 0, stream>>>(tAll, plan);
    k_mix<<<dim3(BB), dim3(256), 0, stream>>>(plan, SB, SA);
  }
  (void)in_sizes; (void)n_in; (void)out_size; (void)ws_size;
}
